// SelfAttention_9405978378594
// MI455X (gfx1250) — matrix operand order repaired and checked by reading, not compiled, not run on hardware
//
#include <hip/hip_runtime.h>
#include <cmath>

typedef __attribute__((ext_vector_type(4)))  unsigned int u32x4;
typedef __attribute__((ext_vector_type(16))) __bf16       bf16x16;
typedef __attribute__((ext_vector_type(8)))  float        f32x8;

struct Frag { u32x4 lo, hi; };

static __device__ inline bf16x16 frag_cast(const Frag& f) {
    return __builtin_bit_cast(bf16x16, f);
}

// round-to-nearest-even f32 -> bf16 (as raw u16)
static __device__ inline unsigned short f2bf(float x) {
    unsigned u = __builtin_bit_cast(unsigned, x);
    u += 0x7FFFu + ((u >> 16) & 1u);
    return (unsigned short)(u >> 16);
}

// ---------------------------------------------------------------- convert
__global__ void cvt_f32_bf16(const float* __restrict__ src,
                             unsigned short* __restrict__ dst, int n) {
    int stride = gridDim.x * blockDim.x;
    for (int i = blockIdx.x * blockDim.x + threadIdx.x; i < n; i += stride)
        dst[i] = f2bf(src[i]);
}

// ---------------------------------------------------------------- GEMM (NT)
// C[m,n] = alpha * sum_k A[m,k] * Bm[n,k]  (+ bias[n])
// A: [M,KC] bf16 row-major, Bm: [N,KC] bf16 row-major.
// KC is a compile-time constant so the 16-row fragment strides (i*16*KC*2
// bytes) become instruction-immediate offsets: one address pointer per
// operand, advanced 64B per K-step -> no in-loop VALU address math, no
// WMMA->VALU hazard NOPs.
// Wave computes a 64x64 output block (4x4 grid of 16x16 WMMA tiles):
// 16 v_wmma per K-step against 8 fragment loads.
#define TM 4
#define TN 4

template <int KC>
__global__ __launch_bounds__(256) void gemm_bf16_nt(
    const unsigned short* __restrict__ A,  long long sAb,
    const unsigned short* __restrict__ Bm, long long sBb,
    const float* __restrict__ bias, float alpha,
    float* __restrict__ Cf, unsigned short* __restrict__ Cb, long long sCb,
    int M, int N)
{
    const int lane = threadIdx.x & 31;
    const int half = lane >> 4;          // 0: lanes 0-15, 1: lanes 16-31
    const int l15  = lane & 15;

    const int wid      = blockIdx.x * (blockDim.x >> 5) + (threadIdx.x >> 5);
    const int n_wtiles = N >> 6;         // 64-wide n blocks
    const int m0 = (wid / n_wtiles) * 64;
    const int n0 = (wid % n_wtiles) * 64;

    const long long b = blockIdx.y;
    // A fragment: lanes 0-15 hold K {0..7, 16..23} of the 32-chunk,
    //             lanes 16-31 hold K {8..15, 24..31}
    const unsigned short* pa = A  + b * sAb + (long long)(m0 + l15) * KC + (half ? 8 : 0);
    // B fragment: lanes 0-15 hold K {0..7, 16..23}, lanes 16-31 hold K {8..15, 24..31}
    const unsigned short* pb = Bm + b * sBb + (long long)(n0 + l15) * KC + (half ? 8 : 0);

    f32x8 acc[TM][TN];
#pragma unroll
    for (int i = 0; i < TM; i++)
#pragma unroll
        for (int j = 0; j < TN; j++) acc[i][j] = (f32x8)(0.0f);

    for (int kk = 0; kk < KC; kk += 32) {
        Frag a[TM], bf[TN];
#pragma unroll
        for (int i = 0; i < TM; i++) {
            a[i].lo = *(const u32x4*)(pa + i * 16 * KC);
            a[i].hi = *(const u32x4*)(pa + i * 16 * KC + 16);
        }
#pragma unroll
        for (int j = 0; j < TN; j++) {
            bf[j].lo = *(const u32x4*)(pb + j * 16 * KC);
            bf[j].hi = *(const u32x4*)(pb + j * 16 * KC + 16);
        }
#pragma unroll
        for (int i = 0; i < TM; i++)
#pragma unroll
            for (int j = 0; j < TN; j++)
                acc[i][j] = __builtin_amdgcn_wmma_f32_16x16x32_bf16(
                    false, frag_cast(a[i]), false, frag_cast(bf[j]),
                    (short)0, acc[i][j], false, false);
        pa += 32;
        pb += 32;
    }

    // C/D layout: lanes 0-15: VGPR v -> M = v, N = lane; lanes 16-31: M = 8+v
#pragma unroll
    for (int j = 0; j < TN; j++) {
        const int col = n0 + 16 * j + l15;
        const float bv = bias ? bias[col] : 0.0f;
#pragma unroll
        for (int i = 0; i < TM; i++) {
            const int rbase = m0 + 16 * i + half * 8;
#pragma unroll
            for (int v = 0; v < 8; v++) {
                float val = acc[i][j][v] * alpha + bv;
                long long idx = b * sCb + (long long)(rbase + v) * N + col;
                if (Cf) Cf[idx] = val;
                if (Cb) Cb[idx] = f2bf(val);
            }
        }
    }
}

// ---------------------------------------------------------------- softmax over QUERY axis
// S: [B, Nq, Nk] fp32.  For each (b,k): m = max_q S, Z = sum_q exp(S-m),
// P[q,k] = exp(S-m)/Z  written as bf16 (this IS attn, 1/Z folded in).
__global__ __launch_bounds__(256) void softmax_col(
    const float* __restrict__ S, unsigned short* __restrict__ P, int Nq, int Nk)
{
    __shared__ float red[256];
    const int b   = blockIdx.y;
    const int k0  = blockIdx.x * 32;
    const int col = threadIdx.x & 31;
    const int r   = threadIdx.x >> 5;
    const float*    Sb = S + (long long)b * Nq * Nk + k0;
    unsigned short* Pb = P + (long long)b * Nq * Nk + k0;

    float pm = -3.4e38f;
    for (int q = r; q < Nq; q += 8)
        pm = fmaxf(pm, Sb[(long long)q * Nk + col]);
    red[threadIdx.x] = pm;
    __syncthreads();
    if (threadIdx.x < 32) {
        float m = red[threadIdx.x];
        for (int i = 1; i < 8; i++) m = fmaxf(m, red[threadIdx.x + 32 * i]);
        red[threadIdx.x] = m;
    }
    __syncthreads();
    const float m = red[col];
    __syncthreads();

    float ps = 0.0f;
    for (int q = r; q < Nq; q += 8)
        ps += __expf(Sb[(long long)q * Nk + col] - m);
    red[threadIdx.x] = ps;
    __syncthreads();
    if (threadIdx.x < 32) {
        float z = red[threadIdx.x];
        for (int i = 1; i < 8; i++) z += red[threadIdx.x + 32 * i];
        red[threadIdx.x] = z;
    }
    __syncthreads();
    const float inv = 1.0f / red[col];

    for (int q = r; q < Nq; q += 8) {
        float e = __expf(Sb[(long long)q * Nk + col] - m) * inv;
        Pb[(long long)q * Nk + col] = f2bf(e);
    }
}

// ---------------------------------------------------------------- bf16 transpose
// src: [b, rows, cols] -> dst: [b, cols, rows]
__global__ __launch_bounds__(256) void transpose_bf16(
    const unsigned short* __restrict__ src, unsigned short* __restrict__ dst,
    int rows, int cols)
{
    __shared__ unsigned short tile[32][33];
    const long long b = blockIdx.z;
    const int r0 = blockIdx.x * 32;
    const int c0 = blockIdx.y * 32;
    const unsigned short* s = src + b * (long long)rows * cols;
    unsigned short*       d = dst + b * (long long)rows * cols;
#pragma unroll
    for (int i = 0; i < 4; i++)
        tile[threadIdx.y + 8 * i][threadIdx.x] =
            s[(long long)(r0 + threadIdx.y + 8 * i) * cols + c0 + threadIdx.x];
    __syncthreads();
#pragma unroll
    for (int i = 0; i < 4; i++)
        d[(long long)(c0 + threadIdx.y + 8 * i) * rows + r0 + threadIdx.x] =
            tile[threadIdx.x][threadIdx.y + 8 * i];
}

// ---------------------------------------------------------------- launch
extern "C" void kernel_launch(void* const* d_in, const int* in_sizes, int n_in,
                              void* d_out, int out_size, void* d_ws, size_t ws_size,
                              hipStream_t stream) {
    (void)in_sizes; (void)n_in; (void)out_size; (void)ws_size;
    const int B = 8, N = 2048, D = 512;
    const long long MN = (long long)B * N;   // 16384 rows total

    const float* x  = (const float*)d_in[0];
    const float* Wq = (const float*)d_in[1];
    const float* bq = (const float*)d_in[2];
    const float* Wk = (const float*)d_in[3];
    const float* bk = (const float*)d_in[4];
    const float* Wv = (const float*)d_in[5];
    const float* bv = (const float*)d_in[6];
    float* out = (float*)d_out;

    char* w = (char*)d_ws;
    size_t off = 0;
    auto alloc = [&](size_t bytes) -> void* {
        void* p = w + off;
        off += (bytes + 255) & ~(size_t)255;
        return p;
    };
    unsigned short* xb  = (unsigned short*)alloc((size_t)MN * D * 2);
    unsigned short* Wqb = (unsigned short*)alloc((size_t)D * D * 2);
    unsigned short* Wkb = (unsigned short*)alloc((size_t)D * D * 2);
    unsigned short* Wvb = (unsigned short*)alloc((size_t)D * D * 2);
    unsigned short* qb  = (unsigned short*)alloc((size_t)MN * D * 2);
    unsigned short* kb  = (unsigned short*)alloc((size_t)MN * D * 2);
    unsigned short* vb  = (unsigned short*)alloc((size_t)MN * D * 2);
    unsigned short* vt  = (unsigned short*)alloc((size_t)MN * D * 2);
    float*          S   = (float*)alloc((size_t)B * N * N * 4);
    unsigned short* P   = (unsigned short*)alloc((size_t)B * N * N * 2);

    // 1) fp32 -> bf16
    cvt_f32_bf16<<<2048, 256, 0, stream>>>(x,  xb,  (int)(MN * D));
    cvt_f32_bf16<<<256,  256, 0, stream>>>(Wq, Wqb, D * D);
    cvt_f32_bf16<<<256,  256, 0, stream>>>(Wk, Wkb, D * D);
    cvt_f32_bf16<<<256,  256, 0, stream>>>(Wv, Wvb, D * D);

    // 2) projections: y = x @ W^T + b   (M=16384, N=512, K=512)
    {
        int M = (int)MN, Nn = D;
        dim3 grid(((M / 64) * (Nn / 64)) / 8, 1);
        gemm_bf16_nt<512><<<grid, 256, 0, stream>>>(xb, 0, Wqb, 0, bq, 1.0f, nullptr, qb, 0, M, Nn);
        gemm_bf16_nt<512><<<grid, 256, 0, stream>>>(xb, 0, Wkb, 0, bk, 1.0f, nullptr, kb, 0, M, Nn);
        gemm_bf16_nt<512><<<grid, 256, 0, stream>>>(xb, 0, Wvb, 0, bv, 1.0f, nullptr, vb, 0, M, Nn);
    }

    // 3) scores: S = (q @ k^T) / (sqrt(D)+1e-6)   per batch  (K=512)
    {
        int M = N, Nn = N;
        dim3 grid(((M / 64) * (Nn / 64)) / 8, B);
        float alpha = 1.0f / (sqrtf((float)D) + 1e-6f);
        gemm_bf16_nt<512><<<grid, 256, 0, stream>>>(qb, (long long)N * D, kb, (long long)N * D,
                                                    nullptr, alpha,
                                                    S, nullptr, (long long)N * N, M, Nn);
    }

    // 4) softmax over the query axis -> P (bf16 attn, 1/Z folded in)
    softmax_col<<<dim3(N / 32, B), 256, 0, stream>>>(S, P, N, N);

    // 5) vt[b,d,k] = vb[b,k,d]
    transpose_bf16<<<dim3(N / 32, D / 32, B), dim3(32, 8), 0, stream>>>(vb, vt, N, D);

    // 6) out = P @ v   (contraction over k; vt rows are over k, K=2048) -> fp32
    {
        int M = N, Nn = D;
        dim3 grid(((M / 64) * (Nn / 64)) / 8, B);
        gemm_bf16_nt<2048><<<grid, 256, 0, stream>>>(P, (long long)N * N, vt, (long long)D * N,
                                                     nullptr, 1.0f,
                                                     out, nullptr, (long long)N * D, M, Nn);
    }
}
